// EGNNLayer_55259049230999
// MI455X (gfx1250) — matrix operand order repaired and checked by reading, not compiled, not run on hardware
//
#include <hip/hip_runtime.h>

// ---------------------------------------------------------------------------
// EGNN layer for MI455X (gfx1250, wave32, WMMA).
// B=2, L=512, FEAT=HID=128.
//
// Decomposition:
//   edge_in @ W_e1 + b_e1 = A[i] + Bn[j] + d2*r      (A folds b_e1, r = W_e1 row 256)
//   t    = silu(...)                  (per-edge 128-vec, built in A-fragment regs)
//   m    = silu(t @ W_e2 + b_e2)      (WMMA f16, f32 acc)
//   m_i  = sum_j pair_mask * m        (reduced straight out of C/D layout)
//   c    = silu(m @ W_c1 + b_c1)      (WMMA, A-frags via LDS transpose stage)
//   w    = tanh(c . W_c2 + b_c2)*0.02, diag-zeroed, pair-masked
//   dx_i = clip(sum_j w * (x_i-x_j)/(dist+eps), +-0.05)
//   h_new = h + silu([h, m_i] @ W_n1 + b_n1) @ W_n2 + b_n2   (cheap scalar kernel)
//
// Round-4 fixes:
//  * split each GEMM's N dimension into two passes of 4 N-tiles: halves the
//    live accumulator set (acc[4]=32 VGPRs instead of acc[8]=64). Goal:
//    total allocation < 256 VGPRs -> no s_set_vgpr_msb SALU tax, scheduler
//    can double-buffer B-fragments (kills per-WMMA s_wait_loadcnt 0x0), and
//    two workgroups fit per WGP (4 waves/SIMD latency hiding).
// ---------------------------------------------------------------------------

#define B_  2
#define L_  512
#define F_  128
#define IB_ 8     // i-nodes per workgroup == waves per workgroup

typedef __attribute__((ext_vector_type(16))) _Float16 v16h;
typedef __attribute__((ext_vector_type(8)))  _Float16 v8h;
typedef __attribute__((ext_vector_type(8)))  float    v8f;

__device__ __forceinline__ float silu_f(float v) {
  // x * sigmoid(x) with v_rcp_f32 (fast path, no div expansion)
  return v * __builtin_amdgcn_rcpf(1.0f + __expf(-v));
}

__device__ __forceinline__ v8f wmma16(v16h a, v16h b, v8f c) {
  // D = A(16x32 f16) * B(32x16 f16) + C(16x16 f32)
  return __builtin_amdgcn_wmma_f32_16x16x32_f16(false, a, false, b, (short)0, c,
                                                false, false);
}

// Opaque zero in an SGPR: defeats LICM hoist-and-spill of "loop-invariant"
// loads while preserving pointer provenance (loads stay global_load/ds_load).
__device__ __forceinline__ int opaque0() {
  int v = 0;
  asm volatile("" : "+s"(v));
  return v;
}

// ---------------------------------------------------------------------------
// Prep 1: A[n,k] = h[n]@W_e1[0:128,k] + b_e1[k] ;  Bn[n,k] = h[n]@W_e1[128:256,k]
// ---------------------------------------------------------------------------
__global__ __launch_bounds__(128) void egnn_prep(
    const float* __restrict__ h, const float* __restrict__ We1,
    const float* __restrict__ be1, float* __restrict__ A,
    float* __restrict__ Bn) {
  __shared__ float sh[F_];
  const int n = blockIdx.x;
  const int t = threadIdx.x;
  sh[t] = h[n * F_ + t];
  __syncthreads();
  float a  = be1[t];
  float bb = 0.0f;
#pragma unroll 4
  for (int f = 0; f < F_; ++f) {
    a  += sh[f] * We1[f * F_ + t];
    bb += sh[f] * We1[(F_ + f) * F_ + t];
  }
  A[n * F_ + t]  = a;
  Bn[n * F_ + t] = bb;
}

// ---------------------------------------------------------------------------
// Prep 2: swizzle W_e2 / W_c1 (K=128 x N=128, f32) into f16 WMMA B-fragment
// order: frag = s*8+nt (K-step s, N-tile nt); within a frag each lane holds 16
// halfs:  element e <-> W[k = 32s + (e<8?0:16) + 8*hi + (e&7)][n = 16*nt + ln].
// ---------------------------------------------------------------------------
__global__ __launch_bounds__(256) void egnn_swz(
    const float* __restrict__ We2, const float* __restrict__ Wc1,
    _Float16* __restrict__ We2sw, _Float16* __restrict__ Wc1sw) {
  const int t     = blockIdx.x * blockDim.x + threadIdx.x;  // 0..32767
  const int which = t >> 14;
  const int idx   = t & 16383;
  const int frag  = idx >> 9;
  const int lane  = (idx >> 4) & 31;
  const int e     = idx & 15;
  const int s  = frag >> 3, nt = frag & 7;
  const int hi = lane >> 4, ln = lane & 15;
  const int k = s * 32 + ((e < 8) ? 0 : 16) + hi * 8 + (e & 7);
  const int n = nt * 16 + ln;
  const float* src = which ? Wc1 : We2;
  _Float16*    dst = which ? Wc1sw : We2sw;
  dst[idx] = (_Float16)src[k * F_ + n];
}

// ---------------------------------------------------------------------------
// Edge kernel: 128 workgroups (b, i-block of 8), 256 threads (8 waves).
// Wave w owns i = I0+w; loop over j-blocks of 16 -> 16x128 A-matrix of edges.
// ---------------------------------------------------------------------------
__global__ __launch_bounds__(256, 1) void egnn_edge(
    const float* __restrict__ A, const float* __restrict__ Bn,
    const _Float16* __restrict__ We2sw, const _Float16* __restrict__ Wc1sw,
    const float* __restrict__ x, const unsigned char* __restrict__ mask,
    const float* __restrict__ We1, const float* __restrict__ be2,
    const float* __restrict__ bc1, const float* __restrict__ Wc2,
    const float* __restrict__ bc2, float* __restrict__ mi,
    float* __restrict__ xout) {
  __shared__ _Float16 sM[IB_][16][F_ + 8];   // m -> A-layout stage (padded)
  __shared__ float    sAi[IB_][F_];
  __shared__ float    sBn[16][F_ + 4];       // padded vs bank conflicts
  __shared__ float    sR[F_];
  __shared__ float    sXj[16][4];
  __shared__ float    sMaskJ[16];
  __shared__ float    sW[IB_][16];

  const int tid  = threadIdx.x;
  const int wave = tid >> 5;
  const int lane = tid & 31;
  const int hi   = lane >> 4;
  const int ln   = lane & 15;
  const int b    = blockIdx.x / (L_ / IB_);
  const int I0   = (blockIdx.x % (L_ / IB_)) * IB_;
  const int i    = I0 + wave;
  const int gi   = b * L_ + i;

  // vectorized staging of per-block i data
  {
    const float4* src = (const float4*)(A + (b * L_ + I0) * F_);
    float4*       dst = (float4*)&sAi[0][0];
    dst[tid] = src[tid];  // 256 float4 == 8*128 floats
  }
  for (int t = tid; t < F_; t += 256) sR[t] = We1[256 * F_ + t];

  const float xi0 = x[gi * 3 + 0], xi1 = x[gi * 3 + 1], xi2 = x[gi * 3 + 2];
  const float maski = mask[gi] ? 1.0f : 0.0f;
  const float bc2s  = bc2[0];

  float be2v[8], bc1v[8], wc2v[8];
#pragma unroll
  for (int nt = 0; nt < 8; ++nt) {
    be2v[nt] = be2[nt * 16 + ln];
    bc1v[nt] = bc1[nt * 16 + ln];
    wc2v[nt] = Wc2[nt * 16 + ln];
  }

  float miacc[8];
#pragma unroll
  for (int nt = 0; nt < 8; ++nt) miacc[nt] = 0.0f;
  float dxa0 = 0.0f, dxa1 = 0.0f, dxa2 = 0.0f;

  for (int J = 0; J < L_; J += 16) {
    __syncthreads();
    {  // stage Bn j-block (16x128 f32) with float4 copies
      const int   row = tid >> 5;        // 0..7
      const int   c0  = (tid & 31);      // 0..31
      const float4* src0 = (const float4*)(Bn + (b * L_ + J) * F_);
      *(float4*)&sBn[row][c0 * 4]     = src0[row * 32 + c0];
      *(float4*)&sBn[row + 8][c0 * 4] = src0[(row + 8) * 32 + c0];
    }
    if (tid < 16) {
      const int gj = b * L_ + J + tid;
      sXj[tid][0] = x[gj * 3 + 0];
      sXj[tid][1] = x[gj * 3 + 1];
      sXj[tid][2] = x[gj * 3 + 2];
      sMaskJ[tid] = mask[gj] ? 1.0f : 0.0f;
    }
    // prefetch next j-block of Bn (8KB = 64 cachelines of 128B)
    if (J + 16 < L_ && tid < 64)
      __builtin_prefetch(Bn + (b * L_ + J + 16) * F_ + tid * 32, 0, 0);
    __syncthreads();

    // per-lane edge (i, J+ln): squared distance (clipped)
    const float dfx = xi0 - sXj[ln][0];
    const float dfy = xi1 - sXj[ln][1];
    const float dfz = xi2 - sXj[ln][2];
    float d2 = dfx * dfx + dfy * dfy + dfz * dfz;
    d2 = fminf(d2, 50.0f);

    // Build A-fragments: t = silu(A_i + Bn_j + d2*r)   (16x32 f16 layout).
    // Laundered bases stop the compiler from pinning the loop-invariant
    // sAi/sR lanes in VGPRs; the k-runs are contiguous -> ds_load_b128.
    const int    lz   = opaque0();
    const float* aRow = sAi[wave] + lz;
    const float* bRow = sBn[ln];
    const float* rRow = sR + lz;
    v16h afrag[4];
#pragma unroll
    for (int s = 0; s < 4; ++s) {
#pragma unroll
      for (int e = 0; e < 16; ++e) {
        const int k = s * 32 + ((e < 8) ? 0 : 16) + 8 * hi + (e & 7);
        afrag[s][e] = (_Float16)silu_f(aRow[k] + bRow[k] + d2 * rRow[k]);
      }
    }

    // GEMM1: m = silu(t @ W_e2 + b_e2)  (32 WMMAs, two N-half passes)
#pragma unroll
    for (int half = 0; half < 2; ++half) {
      v8f acc[4];
#pragma unroll
      for (int q = 0; q < 4; ++q) {
        v8f c;
#pragma unroll
        for (int v = 0; v < 8; ++v) c[v] = be2v[4 * half + q];
        acc[q] = c;
      }
#pragma unroll
      for (int s = 0; s < 4; ++s) {
        const _Float16* wp = We2sw + s * 4096 + half * 2048 + opaque0();
        v16h bf[4];
#pragma unroll
        for (int q = 0; q < 4; ++q)
          bf[q] = *(const v16h*)(wp + (q * 32 + lane) * 16);
#pragma unroll
        for (int q = 0; q < 4; ++q) acc[q] = wmma16(afrag[s], bf[q], acc[q]);
      }
      // silu, stage to LDS (A-layout feed), masked m_i accumulation
#pragma unroll
      for (int q = 0; q < 4; ++q) {
        const int nt = 4 * half + q;
#pragma unroll
        for (int v = 0; v < 8; ++v) {
          const float mval = silu_f(acc[q][v]);
          const int   rr   = v + 8 * hi;  // edge row (j_local)
          sM[wave][rr][nt * 16 + ln] = (_Float16)mval;
          miacc[nt] += mval * sMaskJ[rr];
        }
      }
    }
    __syncthreads();

    // GEMM2: c = silu(m @ W_c1 + b_c1)  (32 WMMAs, two N-half passes)
    // preload all 4 A-fragments from LDS (2x ds_load_b128 each, single wait)
    v16h af2[4];
#pragma unroll
    for (int s = 0; s < 4; ++s) {
      const v8h* pv = (const v8h*)&sM[wave][ln][s * 32 + 8 * hi];
      af2[s] = __builtin_shufflevector(pv[0], pv[2], 0, 1, 2, 3, 4, 5, 6, 7, 8,
                                       9, 10, 11, 12, 13, 14, 15);
    }
    float wpart[8];
#pragma unroll
    for (int v = 0; v < 8; ++v) wpart[v] = 0.0f;
#pragma unroll
    for (int half = 0; half < 2; ++half) {
      v8f acc2[4];
#pragma unroll
      for (int q = 0; q < 4; ++q) {
        v8f c;
#pragma unroll
        for (int v = 0; v < 8; ++v) c[v] = bc1v[4 * half + q];
        acc2[q] = c;
      }
#pragma unroll
      for (int s = 0; s < 4; ++s) {
        const _Float16* wp = Wc1sw + s * 4096 + half * 2048 + opaque0();
        v16h bf[4];
#pragma unroll
        for (int q = 0; q < 4; ++q)
          bf[q] = *(const v16h*)(wp + (q * 32 + lane) * 16);
#pragma unroll
        for (int q = 0; q < 4; ++q) acc2[q] = wmma16(af2[s], bf[q], acc2[q]);
      }
      // w partials: silu(c) . W_c2 (per-row, per-lane-feature slice)
#pragma unroll
      for (int q = 0; q < 4; ++q) {
        const float wc = wc2v[4 * half + q];
#pragma unroll
        for (int v = 0; v < 8; ++v) wpart[v] += silu_f(acc2[q][v]) * wc;
      }
    }

    // w_row = tanh(wpart_sum + b_c2)*0.02, diag-zero, pair-mask
#pragma unroll
    for (int v = 0; v < 8; ++v) {
      float val = wpart[v];
      val += __shfl_xor(val, 8, 16);
      val += __shfl_xor(val, 4, 16);
      val += __shfl_xor(val, 2, 16);
      val += __shfl_xor(val, 1, 16);
      if (ln == v) {
        const int rr = v + 8 * hi;
        float wv = tanhf(val + bc2s) * 0.02f;
        if (J + rr == i) wv = 0.0f;      // zero diagonal
        wv *= maski * sMaskJ[rr];        // pair mask
        sW[wave][rr] = wv;
      }
    }
    __syncthreads();

    // dx accumulation: lanes 0..15 handle row ln
    if (hi == 0) {
      const float wv   = sW[wave][ln];
      const float dist = sqrtf(d2 + 1e-8f);
      const float inv  = __builtin_amdgcn_rcpf(dist + 1e-8f);
      dxa0 += wv * dfx * inv;
      dxa1 += wv * dfy * inv;
      dxa2 += wv * dfz * inv;
    }
  }

  // finalize m_i: combine lane <-> lane+16 halves (row groups), apply mask_i
#pragma unroll
  for (int nt = 0; nt < 8; ++nt) {
    const float other = __shfl_xor(miacc[nt], 16, 32);
    if (hi == 0) mi[gi * F_ + nt * 16 + ln] = (miacc[nt] + other) * maski;
  }

  // finalize dx: reduce 16 lanes, clip, write x_new
#pragma unroll
  for (int off = 8; off >= 1; off >>= 1) {
    dxa0 += __shfl_xor(dxa0, off, 16);
    dxa1 += __shfl_xor(dxa1, off, 16);
    dxa2 += __shfl_xor(dxa2, off, 16);
  }
  if (lane == 0) {
    const float c0 = fminf(fmaxf(dxa0, -0.05f), 0.05f);
    const float c1 = fminf(fmaxf(dxa1, -0.05f), 0.05f);
    const float c2 = fminf(fmaxf(dxa2, -0.05f), 0.05f);
    xout[gi * 3 + 0] = (xi0 + c0) * maski;
    xout[gi * 3 + 1] = (xi1 + c1) * maski;
    xout[gi * 3 + 2] = (xi2 + c2) * maski;
  }
}

// ---------------------------------------------------------------------------
// Node MLP: h_new = h + silu([h, m_i] @ W_n1 + b_n1) @ W_n2 + b_n2
// ---------------------------------------------------------------------------
__global__ __launch_bounds__(128) void egnn_node(
    const float* __restrict__ h, const float* __restrict__ mi,
    const float* __restrict__ Wn1, const float* __restrict__ bn1,
    const float* __restrict__ Wn2, const float* __restrict__ bn2,
    float* __restrict__ hout) {
  __shared__ float sh[F_];
  __shared__ float sm[F_];
  __shared__ float su[F_];
  const int n = blockIdx.x;
  const int t = threadIdx.x;
  sh[t] = h[n * F_ + t];
  sm[t] = mi[n * F_ + t];
  __syncthreads();
  float s = bn1[t];
#pragma unroll 4
  for (int f = 0; f < F_; ++f) {
    s += sh[f] * Wn1[f * F_ + t];
    s += sm[f] * Wn1[(F_ + f) * F_ + t];
  }
  su[t] = silu_f(s);
  __syncthreads();
  float o = bn2[t];
#pragma unroll 4
  for (int f = 0; f < F_; ++f) o += su[f] * Wn2[f * F_ + t];
  hout[n * F_ + t] = sh[t] + o;
}

// ---------------------------------------------------------------------------
extern "C" void kernel_launch(void* const* d_in, const int* in_sizes, int n_in,
                              void* d_out, int out_size, void* d_ws,
                              size_t ws_size, hipStream_t stream) {
  const float*         h    = (const float*)d_in[0];
  const float*         x    = (const float*)d_in[1];
  const unsigned char* mask = (const unsigned char*)d_in[2];  // jax bool -> 1B
  const float* We1 = (const float*)d_in[3];
  const float* be1 = (const float*)d_in[4];
  const float* We2 = (const float*)d_in[5];
  const float* be2 = (const float*)d_in[6];
  const float* Wn1 = (const float*)d_in[7];
  const float* bn1 = (const float*)d_in[8];
  const float* Wn2 = (const float*)d_in[9];
  const float* bn2 = (const float*)d_in[10];
  const float* Wc1 = (const float*)d_in[11];
  const float* bc1 = (const float*)d_in[12];
  const float* Wc2 = (const float*)d_in[13];
  const float* bc2 = (const float*)d_in[14];

  char* ws = (char*)d_ws;
  // workspace layout (1,638,400 bytes total)
  float*    A     = (float*)(ws + 0);        // 2*512*128 f32
  float*    Bn    = (float*)(ws + 524288);   // 2*512*128 f32
  float*    mi    = (float*)(ws + 1048576);  // 2*512*128 f32
  _Float16* We2sw = (_Float16*)(ws + 1572864);  // 16384 f16
  _Float16* Wc1sw = (_Float16*)(ws + 1605632);  // 16384 f16

  float* out  = (float*)d_out;
  float* hout = out;                 // (B,L,128)
  float* xout = out + B_ * L_ * F_;  // (B,L,3)

  egnn_prep<<<B_ * L_, 128, 0, stream>>>(h, We1, be1, A, Bn);
  egnn_swz<<<128, 256, 0, stream>>>(We2, Wc1, We2sw, Wc1sw);
  egnn_edge<<<B_ * (L_ / IB_), 256, 0, stream>>>(A, Bn, We2sw, Wc1sw, x, mask,
                                                 We1, be2, bc1, Wc2, bc2, mi,
                                                 xout);
  egnn_node<<<B_ * L_, 128, 0, stream>>>(h, mi, Wn1, bn1, Wn2, bn2, hout);
}
